// Model_61022895342014
// MI455X (gfx1250) — matrix operand order repaired and checked by reading, not compiled, not run on hardware
//
#include <hip/hip_runtime.h>

// ---------------------------------------------------------------------------
// Fused LSTM (B=512, S=1024, IN=32, H=128) + final linear (OUT=1) for gfx1250.
//
// Grid: 32 blocks (one per 16-row batch tile). Block: 512 threads = 16 wave32.
// Each wave owns 2 N-tiles (32 gate columns) and keeps the f16 B-fragments of
// W_ih^T (K=32) and W_hh^T (K=128, 4 chunks) in VGPRs for the entire sequence.
// Per timestep: 10 x v_wmma_f32_16x16x32_f16 per wave, 2 barriers, gate
// nonlinearities in f32 (raw v_rcp + v_exp), h repacked to f16 A-fragment
// layout in LDS, and the final linear accumulated on the fly (lstm_out never
// touches HBM). x and W_lin loads are software-pipelined one step ahead with
// an additional global_prefetch 8 steps ahead to hide memory latency from the
// sequential recurrence.
// ---------------------------------------------------------------------------

typedef __attribute__((ext_vector_type(16))) _Float16 v16h;
typedef __attribute__((ext_vector_type(4)))  _Float16 v4h;
typedef __attribute__((ext_vector_type(8)))  float    v8f;

#define B_  512
#define S_  1024
#define IN_ 32
#define H_  128
#define G_  512   // 4*H

__device__ __forceinline__ float fast_rcp(float x) {
#if __has_builtin(__builtin_amdgcn_rcpf)
    return __builtin_amdgcn_rcpf(x);     // raw v_rcp_f32, single instruction
#else
    return 1.0f / x;
#endif
}
__device__ __forceinline__ float sigmoid_f(float x) {
    return fast_rcp(1.0f + __expf(-x));
}
__device__ __forceinline__ float tanh_f(float x) {
    float e = __expf(-2.0f * x);
    return (1.0f - e) * fast_rcp(1.0f + e);
}

__global__ __launch_bounds__(512, 1)
void lstm_fused_kernel(const float* __restrict__ x,
                       const float* __restrict__ W_ih,
                       const float* __restrict__ W_hh,
                       const float* __restrict__ b_ih,
                       const float* __restrict__ b_hh,
                       const float* __restrict__ W_lin,
                       const float* __restrict__ b_lin,
                       float* __restrict__ out)
{
    // A-fragment staging buffers (lane-major: [lane][half], 16 halves/lane)
    __shared__ __attribute__((aligned(32))) _Float16 xA[32 * 16];        // x_t  [16x32]
    __shared__ __attribute__((aligned(32))) _Float16 hA[4][32 * 16];     // h    [16x128] in 4 K-chunks
    __shared__ __attribute__((aligned(16))) float    gates[16 * G_];     // pre-activations [16x512]
    __shared__ float biasSum[G_];

    const int tid  = threadIdx.x;
    const int lane = tid & 31;
    const int w    = tid >> 5;          // wave id 0..15
    const int b0   = blockIdx.x * 16;   // first batch row of this WG

    // --- preamble: bias fold (b_ih + b_hh), zero initial h fragments -------
    biasSum[tid] = b_ih[tid] + b_hh[tid];
    #pragma unroll
    for (int q = 0; q < 4; ++q) hA[q][tid] = (_Float16)0.0f;

    // --- preload B fragments (f16) into VGPRs, held for whole sequence -----
    // B-matrix (KxN) 16-bit layout: lane -> column N (=lane%16); K(lane,half) =
    //   (half<8 ? half : half+8) + (lane>=16 ? 8 : 0)
    const int kB = (lane >= 16) ? 8 : 0;
    v16h bih[2];
    v16h bhh[4][2];
    #pragma unroll
    for (int nt = 0; nt < 2; ++nt) {
        const int n = w * 32 + nt * 16 + (lane & 15);   // gate column
        #pragma unroll
        for (int hh = 0; hh < 16; ++hh)
            bih[nt][hh] = (_Float16)W_ih[n * IN_ + (hh + (hh >= 8 ? 8 : 0) + kB)];
        #pragma unroll
        for (int kc = 0; kc < 4; ++kc)
            #pragma unroll
            for (int hh = 0; hh < 16; ++hh)
                bhh[kc][nt][hh] = (_Float16)W_hh[n * H_ + (kc * 32 + hh + (hh >= 8 ? 8 : 0) + kB)];
    }

    // --- x staging map: thread (w, lane) loads x[b0+w, s, lane] ------------
    // A-matrix 16x32 f16 layout: lane holds row M; K(lane,half) =
    //   (half<8 ? half : half+8) + (lane>=16 ? 8 : 0)
    const int xk      = lane;
    const int xs_lane = w + (((xk & 15) >= 8) ? 16 : 0);
    const int xs_half = (xk & 7) + ((xk >= 16) ? 8 : 0);
    const float* xrow = x + (size_t)(b0 + w) * (size_t)S_ * IN_ + xk;

    // --- elementwise map: wave w owns batch row w; lane owns 4 h-columns ---
    const int er  = w;
    const int ej0 = lane * 4;                 // h column base (0..124)
    const int kl0 = ej0 & 31;
    const int ekc = ej0 >> 5;
    const int el  = er + (((kl0 & 15) >= 8) ? 16 : 0);
    const int eh  = (kl0 & 7) + ((kl0 >= 16) ? 8 : 0);
    _Float16* hWr = &hA[ekc][el * 16 + eh];   // 4 consecutive halves (8B aligned)

    float c0 = 0.f, c1 = 0.f, c2 = 0.f, c3 = 0.f;
    float outacc = 0.f;

    const float bz0 = biasSum[w * 32 + (lane & 15)];
    const float bz1 = biasSum[w * 32 + 16 + (lane & 15)];

    // --- software pipeline: prime x and W_lin one step ahead ---------------
    float  xcur  = xrow[0];
    float4 wlcur = *(const float4*)(W_lin + ej0);

    for (int s = 0; s < S_; ++s) {
        // stage x_t (value prefetched last iteration) into A-fragment layout
        xA[xs_lane * 16 + xs_half] = (_Float16)xcur;
        {
            // issue next-step load now; waited one full loop body later
            const int sn = (s + 1 < S_) ? s + 1 : S_ - 1;
            xcur = xrow[(size_t)sn * IN_];
            // warm L2 well ahead of the register pipeline
            const int sp = (s + 8 < S_) ? s + 8 : S_ - 1;
            __builtin_prefetch(&xrow[(size_t)sp * IN_], 0, 1);
        }
        __syncthreads();   // xA ready, hA(s-1) ready, gates(s-1) fully consumed

        // ---- GEMM phase: gates = bias + x_t @ W_ih^T + h @ W_hh^T ----
        v8f acc0, acc1;
        #pragma unroll
        for (int v = 0; v < 8; ++v) { acc0[v] = bz0; acc1[v] = bz1; }

        {
            v16h a = *(const v16h*)&xA[lane * 16];
            acc0 = __builtin_amdgcn_wmma_f32_16x16x32_f16(false, a, false, bih[0], (short)0, acc0, false, false);
            acc1 = __builtin_amdgcn_wmma_f32_16x16x32_f16(false, a, false, bih[1], (short)0, acc1, false, false);
        }
        #pragma unroll
        for (int kc = 0; kc < 4; ++kc) {
            v16h ah = *(const v16h*)&hA[kc][lane * 16];
            acc0 = __builtin_amdgcn_wmma_f32_16x16x32_f16(false, ah, false, bhh[kc][0], (short)0, acc0, false, false);
            acc1 = __builtin_amdgcn_wmma_f32_16x16x32_f16(false, ah, false, bhh[kc][1], (short)0, acc1, false, false);
        }

        // scatter C fragments (lane -> N, vgpr v -> M = v + 8*(lane>=16))
        {
            const int mb = (lane >= 16) ? 8 : 0;
            const int n0 = w * 32 + (lane & 15);
            #pragma unroll
            for (int v = 0; v < 8; ++v) {
                gates[(mb + v) * G_ + n0]      = acc0[v];
                gates[(mb + v) * G_ + n0 + 16] = acc1[v];
            }
        }
        __syncthreads();   // gates visible to elementwise phase

        // ---- elementwise LSTM cell + fused final-linear accumulation ----
        const float* gr = &gates[er * G_ + ej0];
        const float4 gi = *(const float4*)(gr);
        const float4 gf = *(const float4*)(gr + 128);
        const float4 gg = *(const float4*)(gr + 256);
        const float4 go = *(const float4*)(gr + 384);

        const float4 wl = wlcur;
        {
            const int sn = (s + 1 < S_) ? s + 1 : S_ - 1;
            wlcur = *(const float4*)(W_lin + (size_t)sn * H_ + ej0);
        }

        v4h hv;
        {
            float ii = sigmoid_f(gi.x), ff = sigmoid_f(gf.x);
            float gv = tanh_f(gg.x),    oo = sigmoid_f(go.x);
            c0 = ff * c0 + ii * gv;
            float h = oo * tanh_f(c0);
            outacc += h * wl.x; hv[0] = (_Float16)h;
        }
        {
            float ii = sigmoid_f(gi.y), ff = sigmoid_f(gf.y);
            float gv = tanh_f(gg.y),    oo = sigmoid_f(go.y);
            c1 = ff * c1 + ii * gv;
            float h = oo * tanh_f(c1);
            outacc += h * wl.y; hv[1] = (_Float16)h;
        }
        {
            float ii = sigmoid_f(gi.z), ff = sigmoid_f(gf.z);
            float gv = tanh_f(gg.z),    oo = sigmoid_f(go.z);
            c2 = ff * c2 + ii * gv;
            float h = oo * tanh_f(c2);
            outacc += h * wl.z; hv[2] = (_Float16)h;
        }
        {
            float ii = sigmoid_f(gi.w), ff = sigmoid_f(gf.w);
            float gv = tanh_f(gg.w),    oo = sigmoid_f(go.w);
            c3 = ff * c3 + ii * gv;
            float h = oo * tanh_f(c3);
            outacc += h * wl.w; hv[3] = (_Float16)h;
        }
        *(v4h*)hWr = hv;   // repack h into f16 A-fragment layout for next step
    }

    // ---- final reduction: 32 lanes of wave w each hold a partial for row w
    #pragma unroll
    for (int off = 16; off > 0; off >>= 1)
        outacc += __shfl_xor(outacc, off, 32);
    if (lane == 0)
        out[b0 + w] = outacc + b_lin[0];
}

extern "C" void kernel_launch(void* const* d_in, const int* in_sizes, int n_in,
                              void* d_out, int out_size, void* d_ws, size_t ws_size,
                              hipStream_t stream) {
    const float* x     = (const float*)d_in[0];
    const float* W_ih  = (const float*)d_in[1];
    const float* W_hh  = (const float*)d_in[2];
    const float* b_ih  = (const float*)d_in[3];
    const float* b_hh  = (const float*)d_in[4];
    const float* W_lin = (const float*)d_in[5];
    const float* b_lin = (const float*)d_in[6];
    float* out = (float*)d_out;

    (void)in_sizes; (void)n_in; (void)out_size; (void)d_ws; (void)ws_size;

    lstm_fused_kernel<<<dim3(B_ / 16), dim3(512), 0, stream>>>(
        x, W_ih, W_hh, b_ih, b_hh, W_lin, b_lin, out);
}
